// MultiHeadCrossGraphAttention_85856396247185
// MI455X (gfx1250) — matrix operand order repaired and checked by reading, not compiled, not run on hardware
//
#include <hip/hip_runtime.h>
#include <hip/hip_bf16.h>
#include <math.h>

typedef __attribute__((ext_vector_type(16))) _Float16 v16h;
typedef __attribute__((ext_vector_type(8)))  _Float16 v8h;
typedef __attribute__((ext_vector_type(4)))  _Float16 v4h;
typedef __attribute__((ext_vector_type(8)))  float    v8f;
typedef __attribute__((ext_vector_type(4)))  float    v4f;
typedef __attribute__((ext_vector_type(4)))  int      v4i;

#define NQ 4096
#define NK 2048
#define DM 64
#define NH 8

#if __has_builtin(__builtin_amdgcn_global_load_async_to_lds_b128)
#define HAS_ASYNC 1
typedef __attribute__((address_space(1))) v4i* gv4i_p;   // global int4*
typedef __attribute__((address_space(3))) v4i* lv4i_p;   // LDS int4*
#else
#define HAS_ASYNC 0
#endif

__device__ __forceinline__ v16h hcat(v8h lo, v8h hi) {
    return __builtin_shufflevector(lo, hi, 0,1,2,3,4,5,6,7,8,9,10,11,12,13,14,15);
}

__device__ __forceinline__ v8f wmma16(v16h a, v16h b, v8f c) {
    // D = A(16x32 f16) * B(32x16 f16) + C(16x16 f32)
    return __builtin_amdgcn_wmma_f32_16x16x32_f16(false, a, false, b, (short)0, c, false, false);
}

// Load 8 consecutive f32 (16B aligned) and convert to 8 halves.
__device__ __forceinline__ v8h ld8f32h(const float* p) {
    v4f a = *(const v4f*)p;
    v4f b = *(const v4f*)(p + 4);
    v8h r;
#pragma unroll
    for (int i = 0; i < 4; ++i) { r[i] = (_Float16)a[i]; r[i + 4] = (_Float16)b[i]; }
    return r;
}

// A operand (16x32, 16-bit) from f32 row-major [ld=64] matrix, rows rowbase..+15.
// lane m (<16): K = kbase+{0..7,16..23}; lane m+16: K = kbase+{8..15,24..31}.
__device__ __forceinline__ v16h loadA_f32(const float* __restrict__ X, int rowbase,
                                          int kbase, int lane) {
    int m  = lane & 15;
    int lo = kbase + ((lane < 16) ? 0 : 8);
    const float* p = X + (rowbase + m) * DM + lo;
    return hcat(ld8f32h(p), ld8f32h(p + 16));
}

// B operand (32x16, 16-bit) from f16 matrix stored [col j][K] with leading dim ld.
// lanes 0-15 (col j): K = kbase+{0..7,16..23} ; lanes 16-31: K = kbase+{8..15,24..31}.
__device__ __forceinline__ v16h loadB_h16(const _Float16* __restrict__ Wt, int j,
                                          int kbase, int lane, int ld) {
    const _Float16* p = Wt + j * ld + kbase + ((lane < 16) ? 0 : 8);
    return hcat(*(const v8h*)p, *(const v8h*)(p + 16));
}

// ------------------------------------------------ async K-tile staging helpers
__device__ __forceinline__ void stage_k_tile(const _Float16* __restrict__ Kh, int h,
                                             int kc, _Float16* lbuf, int lane) {
    // each lane copies one K row segment (16B): rows kc*32..+31, cols h*8..h*8+7
    const _Float16* g = Kh + (kc * 32 + lane) * DM + h * 8;
    _Float16* l = lbuf + lane * 8;
#if HAS_ASYNC
    __builtin_amdgcn_global_load_async_to_lds_b128((gv4i_p)g, (lv4i_p)l, 0, 0);
#else
    *(v8h*)l = *(const v8h*)g;
#endif
}

__device__ __forceinline__ void wait_async(int more_in_flight) {
#if HAS_ASYNC
#if __has_builtin(__builtin_amdgcn_s_wait_asynccnt)
    if (more_in_flight) __builtin_amdgcn_s_wait_asynccnt(1);
    else                __builtin_amdgcn_s_wait_asynccnt(0);
#else
    if (more_in_flight) asm volatile("s_wait_asynccnt 0x1" ::: "memory");
    else                asm volatile("s_wait_asynccnt 0x0" ::: "memory");
#endif
#endif
}

// ---------------------------------------------------------------- weight prep
__global__ void prep_kernel(const float* __restrict__ wq, const float* __restrict__ wk,
                            const float* __restrict__ wv, const float* __restrict__ wo,
                            _Float16* __restrict__ wtq, _Float16* __restrict__ wtk,
                            _Float16* __restrict__ wtv, _Float16* __restrict__ wto,
                            _Float16* __restrict__ vtp) {
    int b = blockIdx.x, t = threadIdx.x;
    if (b < 4) {
        const float* s = (b == 0) ? wq : (b == 1) ? wk : (b == 2) ? wv : wo;
        _Float16*    d = (b == 0) ? wtq : (b == 1) ? wtk : (b == 2) ? wtv : wto;
        for (int i = t; i < 4096; i += 256) {
            int j = i & 63, k = i >> 6;                 // Wt[j][k] = W[k][j]
            d[j * 64 + k] = (_Float16)s[k * 64 + j];
        }
    } else {
        // zero pad rows d=8..15 of Vt[h][16][NK]
        int base = (b - 4) * 2048;
        for (int i = t; i < 2048; i += 256) {
            int e = base + i;
            int n = e & 2047, rd = e >> 11;             // rd in 0..63
            int h = rd >> 3, d = 8 + (rd & 7);
            vtp[(h * 16 + d) * NK + n] = (_Float16)0.f;
        }
    }
}

// -------------------------------------------- projection GEMM: [N,64]@[64,64]+b
__global__ __launch_bounds__(128) void proj_kernel(
    const float* __restrict__ X, const _Float16* __restrict__ Wt,
    const float* __restrict__ bias, _Float16* __restrict__ outRM,
    _Float16* __restrict__ outVt, float scale, int mode) {
    int lane = threadIdx.x & 31;
    int tile = blockIdx.x * 4 + (threadIdx.x >> 5);
    int rb   = tile * 16;
    v16h a0 = loadA_f32(X, rb, 0, lane);
    v16h a1 = loadA_f32(X, rb, 32, lane);
    int col = lane & 15, lh = lane >> 4;
#pragma unroll
    for (int nt = 0; nt < 4; ++nt) {
        int j = nt * 16 + col;
        v8f acc = {};
        acc = wmma16(a0, loadB_h16(Wt, j, 0, lane, 64), acc);
        acc = wmma16(a1, loadB_h16(Wt, j, 32, lane, 64), acc);
        float bj = bias[j];
#pragma unroll
        for (int r = 0; r < 8; ++r) {
            float v  = (acc[r] + bj) * scale;
            int row  = rb + lh * 8 + r;
            if (mode == 0) {
                outRM[row * DM + j] = (_Float16)v;
            } else {  // V: transposed per-head layout Vt[h][d][n], rows 8..15 pre-zeroed
                int h = j >> 3, d = j & 7;
                outVt[(h * 16 + d) * NK + row] = (_Float16)v;
            }
        }
    }
}

// ------------------------------------------- spatial-MLP bias, tiled f16 cache
// layout: bias[qtile][kchunk][tile2][col16][half2][r8] halves (16B per lane-read)
__global__ __launch_bounds__(128) void bias_kernel(
    const float* __restrict__ qp, const float* __restrict__ kp,
    const float* __restrict__ ws1, const float* __restrict__ bs1,
    const float* __restrict__ ws2, const float* __restrict__ bs2,
    _Float16* __restrict__ biasbuf) {
    int t = threadIdx.x;
    int qtile = blockIdx.x >> 6;
    int kch   = blockIdx.x & 63;
    int e     = t * 4;
    int r0 = e & 7, lh = (e >> 3) & 1, col = (e >> 4) & 15, tl = (e >> 8) & 1;
    int k  = kch * 32 + tl * 16 + col;
    float kx = kp[k * 3 + 0], ky = kp[k * 3 + 1], kz = kp[k * 3 + 2];
    float w1[16], b1[16], w2[16];
#pragma unroll
    for (int i = 0; i < 16; ++i) { w1[i] = ws1[i]; b1[i] = bs1[i]; w2[i] = ws2[i]; }
    float b2 = bs2[0];
    v4h ov;
#pragma unroll
    for (int rr = 0; rr < 4; ++rr) {
        int q = qtile * 16 + lh * 8 + r0 + rr;
        float dx = qp[q * 3 + 0] - kx, dy = qp[q * 3 + 1] - ky, dz = qp[q * 3 + 2] - kz;
        float dist = sqrtf(fmaxf(dx * dx + dy * dy + dz * dz, 0.f));
        float s = b2;
#pragma unroll
        for (int i = 0; i < 16; ++i) s += fmaxf(fmaf(dist, w1[i], b1[i]), 0.f) * w2[i];
        ov[rr] = (_Float16)(1.f / (1.f + __expf(-s)));
    }
    *(v4h*)(biasbuf + ((qtile * 64 + kch) * 2 + tl) * 256 + col * 16 + lh * 8 + r0) = ov;
}

// ---------------------------------------------------------- fused attention
__device__ __forceinline__ void score_tiles(const _Float16* __restrict__ kb,
                                            const _Float16* __restrict__ biasbuf,
                                            v16h aq, int qtile, int kc,
                                            int lane, v8f S[2]) {
    int col = lane & 15, lh = lane >> 4;
#pragma unroll
    for (int t = 0; t < 2; ++t) {
        v8h kr = {};
        if (lane < 16) kr = *(const v8h*)(kb + (t * 16 + col) * 8);   // ds_load_b128
        v8h z = {};
        v8f s0 = {};
        s0 = wmma16(aq, hcat(kr, z), s0);
        const _Float16* bp =
            biasbuf + ((qtile * 64 + kc) * 2 + t) * 256 + col * 16 + lh * 8;
        v8h bb = *(const v8h*)bp;
        __builtin_prefetch(bp + 512, 0, 1);   // next k-chunk's bias tile
#pragma unroll
        for (int r = 0; r < 8; ++r) s0[r] += (float)bb[r];
        S[t] = s0;
    }
}

__global__ __launch_bounds__(128) void attn_kernel(
    const _Float16* __restrict__ Qh, const _Float16* __restrict__ Kh,
    const _Float16* __restrict__ Vt, const _Float16* __restrict__ biasbuf,
    float* __restrict__ attn_out, float* __restrict__ attv) {
    __shared__ alignas(16) _Float16 plds[4][16 * 32];
    __shared__ alignas(16) _Float16 kbuf[4][2][32 * 8];   // double-buffered K stage
    int lane  = threadIdx.x & 31;
    int w     = threadIdx.x >> 5;
    int qtile = blockIdx.x * 4 + w;
    int qb    = qtile * 16;
    int col   = lane & 15, lh = lane >> 4;
    _Float16* pl = plds[w];
    _Float16* kb = &kbuf[w][0][0];

    for (int h = 0; h < NH; ++h) {
        v8h qv = {};
        if (lane < 16) qv = *(const v8h*)(Qh + (qb + col) * DM + h * 8);
        v8h z8 = {};
        v16h aq = hcat(qv, z8);

        float m[8], l[8];
#pragma unroll
        for (int r = 0; r < 8; ++r) { m[r] = -1e30f; l[r] = 0.f; }

        // ---- pass 1: online row max / sum (async double-buffered K tiles)
        stage_k_tile(Kh, h, 0, kb, lane);
        for (int kc = 0; kc < 64; ++kc) {
            int cur = kc & 1;
            if (kc + 1 < 64) stage_k_tile(Kh, h, kc + 1, kb + (cur ^ 1) * 256, lane);
            wait_async(kc + 1 < 64);
            __builtin_amdgcn_wave_barrier();
            v8f S[2];
            score_tiles(kb + cur * 256, biasbuf, aq, qtile, kc, lane, S);
#pragma unroll
            for (int r = 0; r < 8; ++r) {
                float x = fmaxf(S[0][r], S[1][r]);
#pragma unroll
                for (int off = 1; off < 16; off <<= 1) x = fmaxf(x, __shfl_xor(x, off, 32));
                float mn = fmaxf(m[r], x);
                float p  = __expf(S[0][r] - mn) + __expf(S[1][r] - mn);
#pragma unroll
                for (int off = 1; off < 16; off <<= 1) p += __shfl_xor(p, off, 32);
                l[r] = l[r] * __expf(m[r] - mn) + p;
                m[r] = mn;
            }
        }
        float rl[8];
#pragma unroll
        for (int r = 0; r < 8; ++r) rl[r] = 1.f / l[r];

        // ---- pass 2: normalized P -> attn output, P@V via WMMA
        v8f O = {};
        stage_k_tile(Kh, h, 0, kb, lane);
        for (int kc = 0; kc < 64; ++kc) {
            int cur = kc & 1;
            if (kc + 1 < 64) stage_k_tile(Kh, h, kc + 1, kb + (cur ^ 1) * 256, lane);
            wait_async(kc + 1 < 64);
            __builtin_amdgcn_wave_barrier();
            v8f S[2];
            score_tiles(kb + cur * 256, biasbuf, aq, qtile, kc, lane, S);
#pragma unroll
            for (int t = 0; t < 2; ++t) {
#pragma unroll
                for (int r = 0; r < 8; ++r) {
                    float p = __expf(S[t][r] - m[r]) * rl[r];
                    attn_out[(h * NQ + qb + lh * 8 + r) * NK + kc * 32 + t * 16 + col] = p;
                    pl[(lh * 8 + r) * 32 + t * 16 + col] = (_Float16)p;
                }
            }
            __builtin_amdgcn_wave_barrier();
            // re-layout P (C-tile in LDS) -> A operand
            const _Float16* pp = pl + col * 32 + (lh ? 8 : 0);
            v16h ap = hcat(*(const v8h*)pp, *(const v8h*)(pp + 16));
            // V B operand: lane col = output dim d, K = key index
            const _Float16* vp = Vt + (h * 16 + col) * NK + kc * 32 + lh * 8;
            v16h bv = hcat(*(const v8h*)vp, *(const v8h*)(vp + 16));
            O = wmma16(ap, bv, O);
            __builtin_amdgcn_wave_barrier();
        }
        if (col < 8) {
#pragma unroll
            for (int r = 0; r < 8; ++r)
                attv[(qb + lh * 8 + r) * DM + h * 8 + col] = O[r];
        }
    }
}

// ------------------------------- output proj + bias + residual + LayerNorm
__global__ __launch_bounds__(128) void out_kernel(
    const float* __restrict__ attv, const _Float16* __restrict__ Wto,
    const float* __restrict__ bo, const float* __restrict__ qfeat,
    const float* __restrict__ ln_g, const float* __restrict__ ln_b,
    float* __restrict__ out) {
    int lane = threadIdx.x & 31;
    int tile = blockIdx.x * 4 + (threadIdx.x >> 5);
    int rb   = tile * 16;
    v16h a0 = loadA_f32(attv, rb, 0, lane);
    v16h a1 = loadA_f32(attv, rb, 32, lane);
    int col = lane & 15, lh = lane >> 4;
    float x[4][8];
#pragma unroll
    for (int nt = 0; nt < 4; ++nt) {
        int j = nt * 16 + col;
        v8f acc = {};
        acc = wmma16(a0, loadB_h16(Wto, j, 0, lane, 64), acc);
        acc = wmma16(a1, loadB_h16(Wto, j, 32, lane, 64), acc);
        float bj = bo[j];
#pragma unroll
        for (int r = 0; r < 8; ++r)
            x[nt][r] = acc[r] + bj + qfeat[(rb + lh * 8 + r) * DM + j];
    }
#pragma unroll
    for (int r = 0; r < 8; ++r) {
        float s = x[0][r] + x[1][r] + x[2][r] + x[3][r];
#pragma unroll
        for (int off = 1; off < 16; off <<= 1) s += __shfl_xor(s, off, 32);
        float mu = s * (1.f / 64.f);
        float v = 0.f;
#pragma unroll
        for (int nt = 0; nt < 4; ++nt) { float d = x[nt][r] - mu; v += d * d; }
#pragma unroll
        for (int off = 1; off < 16; off <<= 1) v += __shfl_xor(v, off, 32);
        float rstd = rsqrtf(v * (1.f / 64.f) + 1e-5f);
        int row = rb + lh * 8 + r;
#pragma unroll
        for (int nt = 0; nt < 4; ++nt) {
            int j = nt * 16 + col;
            out[row * DM + j] = (x[nt][r] - mu) * rstd * ln_g[j] + ln_b[j];
        }
    }
}

extern "C" void kernel_launch(void* const* d_in, const int* in_sizes, int n_in,
                              void* d_out, int out_size, void* d_ws, size_t ws_size,
                              hipStream_t stream) {
    const float* qf  = (const float*)d_in[0];
    const float* kf  = (const float*)d_in[1];
    const float* vf  = (const float*)d_in[2];
    const float* qp  = (const float*)d_in[3];
    const float* kp  = (const float*)d_in[4];
    const float* wq  = (const float*)d_in[5];
    const float* bq  = (const float*)d_in[6];
    const float* wk  = (const float*)d_in[7];
    const float* bk  = (const float*)d_in[8];
    const float* wv  = (const float*)d_in[9];
    const float* bv  = (const float*)d_in[10];
    const float* wo  = (const float*)d_in[11];
    const float* bo  = (const float*)d_in[12];
    const float* ws1 = (const float*)d_in[13];
    const float* bs1 = (const float*)d_in[14];
    const float* ws2 = (const float*)d_in[15];
    const float* bs2 = (const float*)d_in[16];
    const float* lng = (const float*)d_in[17];
    const float* lnb = (const float*)d_in[18];

    char* ws = (char*)d_ws;
    size_t o = 0;
    _Float16* wtq = (_Float16*)(ws + o); o += 8192;
    _Float16* wtk = (_Float16*)(ws + o); o += 8192;
    _Float16* wtv = (_Float16*)(ws + o); o += 8192;
    _Float16* wto = (_Float16*)(ws + o); o += 8192;
    _Float16* Qh  = (_Float16*)(ws + o); o += (size_t)NQ * DM * 2;
    _Float16* Kh  = (_Float16*)(ws + o); o += (size_t)NK * DM * 2;
    _Float16* Vt  = (_Float16*)(ws + o); o += (size_t)NH * 16 * NK * 2;
    float*    av  = (float*)(ws + o);    o += (size_t)NQ * DM * 4;
    _Float16* bb  = (_Float16*)(ws + o); o += (size_t)NQ * NK * 2;

    float* out  = (float*)d_out;
    float* attn = out + (size_t)NQ * DM;

    prep_kernel<<<68, 256, 0, stream>>>(wq, wk, wv, wo, wtq, wtk, wtv, wto, Vt);
    proj_kernel<<<64, 128, 0, stream>>>(qf, wtq, bq, Qh, nullptr, 0.35355339059f, 0);
    proj_kernel<<<32, 128, 0, stream>>>(kf, wtk, bk, Kh, nullptr, 1.0f, 0);
    proj_kernel<<<32, 128, 0, stream>>>(vf, wtv, bv, nullptr, Vt, 1.0f, 1);
    bias_kernel<<<16384, 128, 0, stream>>>(qp, kp, ws1, bs1, ws2, bs2, bb);
    attn_kernel<<<64, 128, 0, stream>>>(Qh, Kh, Vt, bb, attn, av);
    out_kernel<<<64, 128, 0, stream>>>(av, wto, bo, qf, lng, lnb, out);
}
